// ScriptableConstraintGNN_90202903151104
// MI455X (gfx1250) — matrix operand order repaired and checked by reading, not compiled, not run on hardware
//
#include <hip/hip_runtime.h>
#include <hip/hip_bf16.h>

typedef __attribute__((ext_vector_type(16))) __bf16 v16bf;
typedef __attribute__((ext_vector_type(8)))  float  v8f;

#define HD 64
#define TPW 2        // row tiles per wave in the GEMM
#define LSTRIDE 68   // LDS row stride (floats): 68%64=4 banks shift/row, float4-aligned

// GEMM epilogue modes
#define M_FIRST 0    // first writer of the plane: plain store (replaces zero pass)
#define M_ACC   1    // middle writer: read-modify-write accumulate
#define M_ROOT  2    // last writer: accumulate + bias + ReLU (replaces relu pass)

// node types: 0 ssBox, 1 place_frame, 2 object, 3 ssCylinder, 4 pick, 5 place
__device__ const int g_DST[24] = {0,2,0,1,2,1,5,4,2,0,1,3,4,2,4,1,4,3,5,2,5,3,5,1};

// ---------------- elementwise helper ----------------
__global__ void k_zero(float* __restrict__ p, long long n) {
  long long i = (long long)blockIdx.x * blockDim.x + threadIdx.x;
  if (i < n) p[i] = 0.0f;
}

// ---------------- weight packing into WMMA B-fragment layout ----------------
// out = A @ W^T  =>  B[k][o] = W[o][k].
// B 32x16 fragment, contiguous k as in A: lane L holds column o = L%16; lanes<16 cover K=kstep*32+[0,16),
// lanes>=16 cover K=kstep*32+[16,32); v16bf element e = B[kbase+e][o].
// Packed: [mat][frag=ntile*2+kstep][lane][e] bf16 (16 bf16 = 32B contiguous per lane).
__global__ void k_pack_w(const float* __restrict__ W, __bf16* __restrict__ out, int nmat) {
  int t = blockIdx.x * blockDim.x + threadIdx.x;
  int total = nmat << 12;
  if (t >= total) return;
  int e     = t & 15;
  int lane  = (t >> 4) & 31;
  int frag  = (t >> 9) & 7;
  int mat   = t >> 12;
  int o  = (frag >> 1) * 16 + (lane & 15);
  int kb = (frag & 1) * 32 + ((lane & 16) ? 16 : 0);
  out[t] = (__bf16)W[(size_t)mat * 4096 + (size_t)o * 64 + kb + e];
}

// Fold Wr over edge types sharing the same dst: out[mat=l*6+d] = sum_{i:DST[i]==d} Wr[l*24+i],
// packed into the same B-fragment layout (fold in f32, convert once).
__global__ void k_pack_wsum(const float* __restrict__ W, __bf16* __restrict__ out) {
  int t = blockIdx.x * blockDim.x + threadIdx.x;
  int total = 18 << 12;                        // 3 layers * 6 dst types
  if (t >= total) return;
  int e     = t & 15;
  int lane  = (t >> 4) & 31;
  int frag  = (t >> 9) & 7;
  int mat   = t >> 12;
  int l = mat / 6, d = mat % 6;
  int o  = (frag >> 1) * 16 + (lane & 15);
  int kb = (frag & 1) * 32 + ((lane & 16) ? 16 : 0);
  float s = 0.0f;
  for (int i = 0; i < 24; ++i)
    if (g_DST[i] == d)
      s += W[(size_t)(l * 24 + i) * 4096 + (size_t)o * 64 + kb + e];
  out[t] = (__bf16)s;
}

// Fold bl the same way: out[l][d][o] = sum_{i:DST[i]==d} bl[l*24+i][o]
__global__ void k_bias_sum(const float* __restrict__ bl, float* __restrict__ out) {
  int t = blockIdx.x * blockDim.x + threadIdx.x;
  if (t >= 3 * 6 * HD) return;
  int o = t & 63;
  int d = (t >> 6) % 6;
  int l = t / (6 * HD);
  float s = 0.0f;
  for (int i = 0; i < 24; ++i)
    if (g_DST[i] == d) s += bl[(size_t)(l * 24 + i) * HD + o];
  out[t] = s;
}

// ---------------- initial node embedding (feat ++ sinusoidal PE) ----------------
// div = [1, 0.01]  =>  pe(t) = [sin t, cos t, sin .01t, cos .01t]
__global__ void k_embed(const float* __restrict__ x, int dinf,
                        const int* __restrict__ times,
                        const float* __restrict__ W, const float* __restrict__ b,
                        float* __restrict__ out, int N) {
  int t = blockIdx.x * blockDim.x + threadIdx.x;
  if (t >= N * HD) return;
  int n = t >> 6, o = t & 63;
  int din = dinf + 4;
  const float* wr = W + (size_t)o * din;
  float acc = b[o];
  for (int k = 0; k < dinf; ++k) acc += x[(size_t)n * dinf + k] * wr[k];
  float tt = (float)times[n];
  acc += sinf(tt)         * wr[dinf + 0] + cosf(tt)         * wr[dinf + 1]
       + sinf(tt * 0.01f) * wr[dinf + 2] + cosf(tt * 0.01f) * wr[dinf + 3];
  out[t] = fmaxf(acc, 0.0f);
}

// ---------------- edge scatter: sums[d] += h[s], deg[d] += 1 ----------------
__global__ void k_scatter(const float* __restrict__ hsrc, const int* __restrict__ es,
                          const int* __restrict__ ed, float* __restrict__ sums,
                          float* __restrict__ deg, int E) {
  int t = blockIdx.x * blockDim.x + threadIdx.x;
  int e = t >> 6, f = t & 63;
  if (e >= E) return;
  int s = es[e], d = ed[e];
  __hip_atomic_fetch_add(&sums[(size_t)d * HD + f], hsrc[(size_t)s * HD + f],
                         __ATOMIC_RELAXED, __HIP_MEMORY_SCOPE_AGENT);
  if (f == 0)
    __hip_atomic_fetch_add(&deg[d], 1.0f, __ATOMIC_RELAXED, __HIP_MEMORY_SCOPE_AGENT);
}

// ---------------- A-fragment load (f32 -> bf16, contiguous-k A 16x32 layout) ----------------
// Lane L: row m = m0 + L%16; kb = kstep*32 + (L<16?0:16); elements 0..7 <- A[m][kb..kb+7],
// elements 8..15 <- A[m][kb+8..kb+15].
__device__ __forceinline__ v16bf load_a_frag(const float* __restrict__ base,
                                             int row, int kb, float scale) {
  const float4* p0 = reinterpret_cast<const float4*>(base + (size_t)row * HD + kb);
  const float4* p1 = reinterpret_cast<const float4*>(base + (size_t)row * HD + kb + 8);
  float4 a0 = p0[0], a1 = p0[1], a2 = p1[0], a3 = p1[1];
  v16bf r;
  r[0]  = (__bf16)(a0.x * scale); r[1]  = (__bf16)(a0.y * scale);
  r[2]  = (__bf16)(a0.z * scale); r[3]  = (__bf16)(a0.w * scale);
  r[4]  = (__bf16)(a1.x * scale); r[5]  = (__bf16)(a1.y * scale);
  r[6]  = (__bf16)(a1.z * scale); r[7]  = (__bf16)(a1.w * scale);
  r[8]  = (__bf16)(a2.x * scale); r[9]  = (__bf16)(a2.y * scale);
  r[10] = (__bf16)(a2.z * scale); r[11] = (__bf16)(a2.w * scale);
  r[12] = (__bf16)(a3.x * scale); r[13] = (__bf16)(a3.y * scale);
  r[14] = (__bf16)(a3.z * scale); r[15] = (__bf16)(a3.w * scale);
  return r;
}

// ---------------- GEMM: out (+)= A' @ W^T, A' = A/deg (agg) or A (root) ----------------
// MODE = M_FIRST: agg GEMM, first writer -> plain store (no zero pass needed)
// MODE = M_ACC  : agg GEMM, accumulate into out
// MODE = M_ROOT : root GEMM, accumulate + bias + ReLU (last writer -> no relu pass needed)
// Np is a multiple of 16 and all row buffers have Np rows: full tiles, no masks, EXEC
// all-ones throughout. Block stages the 8KB packed weight matrix in LDS; each wave holds
// its 8 B-fragments in registers and sweeps TPW 16-row tiles: A-load clause -> 8
// back-to-back WMMAs -> accumulator transpose through a private LDS patch -> coalesced
// float4 store / read-modify-write of out.
template <int MODE>
__global__ void __launch_bounds__(256)
__attribute__((amdgpu_waves_per_eu(1)))
k_gemm(const float* __restrict__ A, const float* __restrict__ deg,
       const __bf16* __restrict__ wpk, const float* __restrict__ bias,
       float* __restrict__ out, int Np) {
  __shared__ __attribute__((aligned(32))) __bf16 lB[4096];
  __shared__ __attribute__((aligned(16))) float lT[8 * 16 * LSTRIDE];
  reinterpret_cast<v16bf*>(lB)[threadIdx.x] =
      reinterpret_cast<const v16bf*>(wpk)[threadIdx.x];
  __syncthreads();

  int lane = threadIdx.x & 31;
  int wave = threadIdx.x >> 5;
  const v16bf* Lp = reinterpret_cast<const v16bf*>(lB);
  v16bf bf[8];
#pragma unroll
  for (int f = 0; f < 8; ++f) bf[f] = Lp[f * 32 + lane];   // frag = ntile*2 + kstep

  int ncol = lane & 15;
  int koff = (lane & 16) ? 16 : 0;
  int hi8  = (lane & 16) ? 8 : 0;
  float* lrow = lT + wave * (16 * LSTRIDE);
  int tile0 = (blockIdx.x * 8 + wave) * TPW;

#pragma unroll
  for (int tt = 0; tt < TPW; ++tt) {
    int m0 = (tile0 + tt) * 16;
    if (m0 >= Np) break;                     // wave-uniform
    int mr = m0 + (lane & 15);
    float scale = (MODE == M_ROOT) ? 1.0f : (1.0f / fmaxf(deg[mr], 1.0f));

    v16bf a0 = load_a_frag(A, mr, koff, scale);        // K = [0,32)
    v16bf a1 = load_a_frag(A, mr, 32 + koff, scale);   // K = [32,64)

    v8f acc[4] = {};
#pragma unroll
    for (int nt = 0; nt < 4; ++nt) {
      acc[nt] = __builtin_amdgcn_wmma_f32_16x16x32_bf16(
          false, a0, false, bf[nt * 2 + 0], (short)0, acc[nt], false, false);
      acc[nt] = __builtin_amdgcn_wmma_f32_16x16x32_bf16(
          false, a1, false, bf[nt * 2 + 1], (short)0, acc[nt], false, false);
    }

    // D layout: VGPR r -> row hi8+r, column lane%16 + 16*ntile. Transpose via LDS patch.
    // Bias add only exists in the ROOT instantiation (x+0.0f is not an FP no-op).
#pragma unroll
    for (int nt = 0; nt < 4; ++nt) {
      float bv = (MODE == M_ROOT) ? bias[nt * 16 + ncol] : 0.0f;
#pragma unroll
      for (int r = 0; r < 8; ++r) {
        float val = acc[nt][r];
        if (MODE == M_ROOT) val += bv;
        lrow[(hi8 + r) * LSTRIDE + nt * 16 + ncol] = val;
      }
    }
    // Same-wave DS ordering: no barrier needed. Read back row-major, wide store/RMW.
    int rrow = lane & 15;
    int ch   = lane >> 4;                    // column half: 32 floats
    float* gp = out + (size_t)(m0 + rrow) * HD + ch * 32;
    const float* lp = lrow + rrow * LSTRIDE + ch * 32;
#pragma unroll
    for (int j = 0; j < 8; ++j) {
      float4 v = reinterpret_cast<const float4*>(lp)[j];
      if (MODE == M_FIRST) {
        reinterpret_cast<float4*>(gp)[j] = v;
      } else {
        float4 g = reinterpret_cast<float4*>(gp)[j];
        g.x += v.x; g.y += v.y; g.z += v.z; g.w += v.w;
        if (MODE == M_ROOT) {
          g.x = fmaxf(g.x, 0.0f); g.y = fmaxf(g.y, 0.0f);
          g.z = fmaxf(g.z, 0.0f); g.w = fmaxf(g.w, 0.0f);
        }
        reinterpret_cast<float4*>(gp)[j] = g;
      }
    }
  }
}

// ---------------- output head: relu(z @ W1^T + b1) @ W2^T + b2 ----------------
__global__ void k_head(const float* __restrict__ hpick, const float* __restrict__ hplace,
                       const float* __restrict__ W1, const float* __restrict__ b1,
                       const float* __restrict__ W2, const float* __restrict__ b2,
                       float* __restrict__ out, int N) {
  int r = blockIdx.x * blockDim.x + threadIdx.x;
  if (r >= 2 * N) return;
  const float* z = (r < N) ? (hpick + (size_t)r * HD) : (hplace + (size_t)(r - N) * HD);
  float res = b2[0];
  for (int j = 0; j < 16; ++j) {
    float a = b1[j];
#pragma unroll 8
    for (int k = 0; k < HD; ++k) a += z[k] * W1[j * HD + k];
    res += fmaxf(a, 0.0f) * W2[j];
  }
  out[r] = res;
}

// ---------------- host ----------------
static inline size_t align256(size_t x) { return (x + 255) & ~(size_t)255; }

extern "C" void kernel_launch(void* const* d_in, const int* in_sizes, int n_in,
                              void* d_out, int out_size, void* d_ws, size_t ws_size,
                              hipStream_t stream) {
  (void)n_in; (void)out_size; (void)ws_size;
  const float* x_feat[4] = {(const float*)d_in[0], (const float*)d_in[1],
                            (const float*)d_in[2], (const float*)d_in[3]};
  const int* times[6];
  for (int i = 0; i < 6; ++i) times[i] = (const int*)d_in[4 + i];
  const int* ei = (const int*)d_in[10];
  int N = in_sizes[4];
  int E = in_sizes[10] / 48;
  const float *Wn[6], *bn[6];
  for (int i = 0; i < 6; ++i) {
    Wn[i] = (const float*)d_in[13 + 2 * i];
    bn[i] = (const float*)d_in[14 + 2 * i];
  }
  const float* Wl = (const float*)d_in[25];
  const float* bl = (const float*)d_in[26];
  const float* Wr = (const float*)d_in[27];
  const float* W1 = (const float*)d_in[28];
  const float* b1 = (const float*)d_in[29];
  const float* W2 = (const float*)d_in[30];
  const float* b2 = (const float*)d_in[31];

  // node types: 0 ssBox, 1 place_frame, 2 object, 3 ssCylinder, 4 pick, 5 place
  static const int SRC[24] = {2,0,1,0,1,2,4,5,2,0,1,3,2,4,1,4,3,4,2,5,3,5,1,5};
  static const int DST[24] = {0,2,0,1,2,1,5,4,2,0,1,3,4,2,4,1,4,3,5,2,5,3,5,1};
  static const int DINF[6] = {4,4,4,3,0,0};

  int Np = (N + 15) & ~15;                 // pad rows to full 16-row tiles
  size_t nhp = (size_t)Np * HD;
  char* w = (char*)d_ws;
  size_t off = 0;
  float* h_cur = (float*)(w + off); off = align256(off + 6 * nhp * sizeof(float));
  float* h_nxt = (float*)(w + off); off = align256(off + 6 * nhp * sizeof(float));
  float* sums  = (float*)(w + off); off += nhp * sizeof(float);
  float* deg   = (float*)(w + off); off = align256(off + (size_t)Np * sizeof(float));
  __bf16* wlpk = (__bf16*)(w + off); off = align256(off + (size_t)72 * 4096 * sizeof(__bf16));
  __bf16* wspk = (__bf16*)(w + off); off = align256(off + (size_t)18 * 4096 * sizeof(__bf16));
  float* bsum  = (float*)(w + off); off = align256(off + (size_t)3 * 6 * HD * sizeof(float));

  const int TB = 256;

  // pack the 72 Wl matrices; fold+pack Wr into 18 per-(layer,dst) matrices; fold biases
  {
    int total = 72 << 12;
    k_pack_w<<<(total + TB - 1) / TB, TB, 0, stream>>>(Wl, wlpk, 72);
    int total2 = 18 << 12;
    k_pack_wsum<<<(total2 + TB - 1) / TB, TB, 0, stream>>>(Wr, wspk);
    k_bias_sum<<<(3 * 6 * HD + TB - 1) / TB, TB, 0, stream>>>(bl, bsum);
  }

  long long tot6 = (long long)6 * nhp;
  int g6 = (int)((tot6 + TB - 1) / TB);

  // initial embeddings (zero first so pad rows are deterministic zeros)
  {
    k_zero<<<g6, TB, 0, stream>>>(h_cur, tot6);
    long long tot = (long long)N * HD;
    int g = (int)((tot + TB - 1) / TB);
    for (int t = 0; t < 6; ++t) {
      const float* xp = (t < 4) ? x_feat[t] : nullptr;
      k_embed<<<g, TB, 0, stream>>>(xp, DINF[t], times[t], Wn[t], bn[t],
                                    h_cur + (size_t)t * nhp, N);
    }
  }

  long long totsd = (long long)(nhp + (size_t)Np);
  int gsd = (int)((totsd + TB - 1) / TB);
  int gscat = (int)(((long long)E * HD + TB - 1) / TB);
  int tiles = Np / 16;
  int ggemm = (tiles + 8 * TPW - 1) / (8 * TPW);

  float* hc = h_cur;
  float* hn = h_nxt;
  for (int l = 0; l < 3; ++l) {
    bool written[6] = {false, false, false, false, false, false};
    // neighbor aggregation + Wl transform per edge type; first writer per plane stores
    for (int i = 0; i < 24; ++i) {
      k_zero<<<gsd, TB, 0, stream>>>(sums, totsd);  // sums + contiguous deg
      const int* es = ei + (size_t)i * 2 * E;
      const int* ed = es + E;
      k_scatter<<<gscat, TB, 0, stream>>>(hc + (size_t)SRC[i] * nhp, es, ed, sums, deg, E);
      int mi = l * 24 + i;
      float* outp = hn + (size_t)DST[i] * nhp;
      const __bf16* wp = wlpk + (size_t)mi * 4096;
      if (!written[DST[i]]) {
        k_gemm<M_FIRST><<<ggemm, 256, 0, stream>>>(sums, deg, wp, nullptr, outp, Np);
        written[DST[i]] = true;
      } else {
        k_gemm<M_ACC><<<ggemm, 256, 0, stream>>>(sums, deg, wp, nullptr, outp, Np);
      }
    }
    // folded root transform (last writer per plane): accumulate + bias + ReLU
    for (int t = 0; t < 6; ++t) {
      int mi = l * 6 + t;
      k_gemm<M_ROOT><<<ggemm, 256, 0, stream>>>(hc + (size_t)t * nhp, nullptr,
                                                wspk + (size_t)mi * 4096,
                                                bsum + (size_t)mi * HD,
                                                hn + (size_t)t * nhp, Np);
    }
    float* tmp = hc; hc = hn; hn = tmp;
  }

  int g2n = (2 * N + TB - 1) / TB;
  k_head<<<g2n, TB, 0, stream>>>(hc + (size_t)4 * nhp, hc + (size_t)5 * nhp,
                                 W1, b1, W2, b2, (float*)d_out, N);
}
